// CosAttn2d_22857815949317
// MI455X (gfx1250) — matrix operand order repaired and checked by reading, not compiled, not run on hardware
//
#include <hip/hip_runtime.h>

typedef __attribute__((ext_vector_type(16))) __bf16 v16bf;
typedef __attribute__((ext_vector_type(8)))  float  v8f;
typedef __attribute__((ext_vector_type(4)))  unsigned int u32x4;

union FragBF {
    v16bf v;
    unsigned int   u32[8];
    unsigned short u16[16];
};

#define DN_CONST   0.21022410381342863f   // 512^-0.25
#define FREQ_CONST 0.032724923474893676f  // pi / 96

// LDS layout (bytes); chunk = 128 spatial columns
#define OFF_TRIG 0u        // 32*2 floats        = 256 B
#define OFF_KSUM 256u      // 4*64 floats        = 1024 B
#define OFF_DINV 1280u     // 128 floats         = 512 B
#define OFF_KAX  1792u     // 2 x [64][128] bf16 = 32768 B (copy0 reused for qa in ph2)
#define OFF_VT   34560u    // [128][64] bf16     = 16384 B (transposed v)
#define OFF_CTX  50944u    // 4*64*64 f32        = 65536 B
#define SMEM_BYTES 116480u

__device__ __forceinline__ unsigned short f2bf(float f) {
    unsigned int u = __builtin_bit_cast(unsigned int, f);
    return (unsigned short)((u + 0x7FFFu + ((u >> 16) & 1u)) >> 16);
}
__device__ __forceinline__ float bf2f(unsigned short h) {
    unsigned int u = ((unsigned int)h) << 16;
    return __builtin_bit_cast(float, u);
}
__device__ __forceinline__ unsigned int pack2(float a, float b) {
    return (unsigned int)f2bf(a) | ((unsigned int)f2bf(b) << 16);
}

// K-position of A-fragment element e (16-bit A 16x32 layout, ISA 7.12.2)
__device__ __forceinline__ int a_kmap(int e, int hi) {
    int i = e >> 1;
    int base = (i < 4) ? (2 * i) : (16 + 2 * (i - 4));
    return base + 8 * hi + (e & 1);
}

__global__ __launch_bounds__(256)
void cosattn2d_kernel(const float* __restrict__ qg,
                      const float* __restrict__ kg,
                      const float* __restrict__ vg,
                      float* __restrict__ outg) {
    extern __shared__ char sm[];
    float*          trig = (float*)(sm + OFF_TRIG);          // [32][2] cos,sin
    float*          ksum = (float*)(sm + OFF_KSUM);          // [4][64]
    float*          dinv = (float*)(sm + OFF_DINV);          // [128]
    unsigned short* kaX  = (unsigned short*)(sm + OFF_KAX);  // [2][64][128] (k*tx_cx), ph2: qa
    unsigned short* vT   = (unsigned short*)(sm + OFF_VT);   // [128][64] transposed v
    float*          ctx  = (float*)(sm + OFF_CTX);           // [4][64][64]

    const int tid  = threadIdx.x;
    const int wv   = tid >> 5;
    const int lane = tid & 31;
    const int lo   = lane & 15;
    const int hi   = lane >> 4;
    const size_t base = (size_t)blockIdx.x * 64u * 1024u;    // (b*8+h)*64*1024

    // ---- init: trig table + ksum zero ----
    if (tid < 32) {
        float s, c;
        __sincosf((float)tid * FREQ_CONST, &s, &c);
        trig[tid * 2 + 0] = c;
        trig[tid * 2 + 1] = s;
    }
    ksum[tid] = 0.0f;   // 4*64 == 256 == blockDim
    __syncthreads();

    // ================= Phase 1: ctx_{2cy+cx} = sum_y ty_cy(y) * (ka*tx_cx) @ vT =================
    const int cx   = wv & 1;        // x-trig selector for this wave (0=cos,1=sin)
    const int dh   = wv >> 1;       // d1 quarter (0..3)
    const int d1_0 = dh * 16;

    v8f acc[2][4];                   // [cy][d2 tile]
    #pragma unroll
    for (int a = 0; a < 2; ++a)
        #pragma unroll
        for (int b = 0; b < 4; ++b)
            #pragma unroll
            for (int r = 0; r < 8; ++r) acc[a][b][r] = 0.0f;

    v8f zc;
    #pragma unroll
    for (int r = 0; r < 8; ++r) zc[r] = 0.0f;

    float kpart[4] = {0.f, 0.f, 0.f, 0.f};
    const int rowS = tid >> 2;          // staging row 0..63
    const int cseg = (tid & 3) * 32;    // staging col base (one image row each)

    for (int ch = 0; ch < 8; ++ch) {
        const int ncb = ch * 128;
        // ---- stage k*txc, k*txs, vT to LDS as bf16; fold ksum ----
        {
            const float* kp = kg + base + (size_t)rowS * 1024 + ncb + cseg;
            const float* vp = vg + base + (size_t)rowS * 1024 + ncb + cseg;
            unsigned int* kd0 = (unsigned int*)(kaX + rowS * 128 + cseg);
            unsigned int* kd1 = (unsigned int*)(kaX + 64 * 128 + rowS * 128 + cseg);
            const int y = (ncb + cseg) >> 5;        // constant per thread per chunk
            const float tyc = trig[y * 2], tys = trig[y * 2 + 1];
            float p0 = 0.f, p1 = 0.f;               // sum k*txc, sum k*txs
            #pragma unroll 4
            for (int j = 0; j < 32; j += 2) {
                float2 kf = *(const float2*)(kp + j);
                float2 vf = *(const float2*)(vp + j);
                float k0 = fmaxf(kf.x * DN_CONST, 0.f) + 1e-5f;
                float k1 = fmaxf(kf.y * DN_CONST, 0.f) + 1e-5f;
                float xc0 = trig[j * 2],       xs0 = trig[j * 2 + 1];
                float xc1 = trig[(j + 1) * 2], xs1 = trig[(j + 1) * 2 + 3 - 2];
                float kc0 = k0 * xc0, kc1 = k1 * xc1;
                float ks0 = k0 * xs0, ks1 = k1 * xs1;
                kd0[j >> 1] = pack2(kc0, kc1);
                kd1[j >> 1] = pack2(ks0, ks1);
                p0 += kc0 + kc1;
                p1 += ks0 + ks1;
                // transposed v store: vT[nloc][d]
                vT[(cseg + j) * 64 + rowS]     = f2bf(vf.x);
                vT[(cseg + j + 1) * 64 + rowS] = f2bf(vf.y);
            }
            kpart[0] += tyc * p0;  kpart[1] += tyc * p1;
            kpart[2] += tys * p0;  kpart[3] += tys * p1;
        }
        __syncthreads();

        // ---- WMMA: 4 K-steps of 32 (one image row each); A via b128 loads, B via strided u16 loads ----
        #pragma unroll
        for (int ks = 0; ks < 4; ++ks) {
            const int n0 = ks * 32;
            const int y  = ch * 4 + ks;
            const float tyc = trig[y * 2], tys = trig[y * 2 + 1];

            FragBF a;
            {
                const unsigned short* src = kaX + cx * (64 * 128) + (d1_0 + lo) * 128 + n0 + hi * 8;
                ((u32x4*)a.u32)[0] = *(const u32x4*)(src);
                ((u32x4*)a.u32)[1] = *(const u32x4*)(src + 16);
            }
            #pragma unroll
            for (int dt = 0; dt < 4; ++dt) {
                FragBF b;   // B[kk,nn] = vT[n0+kk][dt*16+nn], nn = lo, kk = a_kmap(e, hi)
                const unsigned short* src = vT + n0 * 64 + dt * 16 + lo;
                #pragma unroll
                for (int e = 0; e < 16; ++e) b.u16[e] = src[a_kmap(e, hi) * 64];
                v8f t = __builtin_amdgcn_wmma_f32_16x16x32_bf16(
                            false, a.v, false, b.v, (short)0, zc, false, false);
                #pragma unroll
                for (int r = 0; r < 8; ++r) {
                    acc[0][dt][r] += tyc * t[r];
                    acc[1][dt][r] += tys * t[r];
                }
            }
        }
        __syncthreads();
    }

    // ---- spill ctx tiles (c = 2*cy + cx) to LDS, commit ksum ----
    #pragma unroll
    for (int cy = 0; cy < 2; ++cy)
        #pragma unroll
        for (int dt = 0; dt < 4; ++dt)
            #pragma unroll
            for (int r = 0; r < 8; ++r) {
                int c    = (cy << 1) | cx;
                int drow = d1_0 + r + 8 * hi;
                int ecol = dt * 16 + lo;
                ctx[(c * 64 + drow) * 64 + ecol] = acc[cy][dt][r];
            }
    #pragma unroll
    for (int c = 0; c < 4; ++c) atomicAdd(&ksum[c * 64 + rowS], kpart[c]);
    __syncthreads();

    // ================= Phase 2: out^T = sum_c coef_c(n) * ctx_c^T @ qa^T, /D =================
    const int e0  = (wv & 3) * 16;
    const int nh4 = (wv >> 2) * 4;

    // per-wave cached A fragments: A[m=e, kk=dd] = ctx_c[dd, e0+m]
    FragBF af[4][2];
    #pragma unroll
    for (int c = 0; c < 4; ++c)
        #pragma unroll
        for (int kst = 0; kst < 2; ++kst)
            #pragma unroll
            for (int e = 0; e < 16; ++e) {
                int dd = kst * 32 + a_kmap(e, hi);
                af[c][kst].u16[e] = f2bf(ctx[(c * 64 + dd) * 64 + e0 + lo]);
            }

    for (int ch = 0; ch < 8; ++ch) {
        const int ncb = ch * 128;
        // ---- stage qa chunk (reuse kaX copy 0) ----
        {
            const float* qp = qg + base + (size_t)rowS * 1024 + ncb + cseg;
            unsigned int* qd = (unsigned int*)(kaX + rowS * 128 + cseg);
            #pragma unroll 4
            for (int j = 0; j < 32; j += 2) {
                float2 qf2 = *(const float2*)(qp + j);
                float q0 = fmaxf(qf2.x * DN_CONST, 0.f) + 1e-5f;
                float q1 = fmaxf(qf2.y * DN_CONST, 0.f) + 1e-5f;
                qd[j >> 1] = pack2(q0, q1);
            }
        }
        __syncthreads();
        // ---- D(n) and 1/D for this chunk's 128 columns ----
        if (tid < 128) {
            int n = ncb + tid;
            int y = n >> 5, x = n & 31;
            float s0 = 0.f, s1 = 0.f, s2 = 0.f, s3 = 0.f;
            #pragma unroll 8
            for (int d = 0; d < 64; ++d) {
                float qv = bf2f(kaX[d * 128 + tid]);
                s0 += qv * ksum[d];
                s1 += qv * ksum[64 + d];
                s2 += qv * ksum[128 + d];
                s3 += qv * ksum[192 + d];
            }
            float tyc = trig[y * 2], tys = trig[y * 2 + 1];
            float txc = trig[x * 2], txs = trig[x * 2 + 1];
            float Dv = tyc * txc * s0 + tyc * txs * s1 + tys * txc * s2 + tys * txs * s3;
            dinv[tid] = 1.0f / Dv;
        }
        __syncthreads();
        // ---- WMMA output tiles: wave covers e-block e0, n-blocks nh4..nh4+3 ----
        #pragma unroll
        for (int jj = 0; jj < 4; ++jj) {
            const int nbl = (nh4 + jj) * 16;
            FragBF b[2];   // B[kk,nn] = qa[kst*32+kk, nbl+nn], nn = lo, kk = a_kmap(e, hi)
            #pragma unroll
            for (int kst = 0; kst < 2; ++kst) {
                const unsigned short* src = kaX + (kst * 32) * 128 + nbl + lo;
                #pragma unroll
                for (int e = 0; e < 16; ++e) b[kst].u16[e] = src[a_kmap(e, hi) * 128];
            }
            const int n_lane = ncb + nbl + lo;
            const int y = n_lane >> 5, x = n_lane & 31;
            const float dv = dinv[nbl + lo];

            v8f oacc;
            #pragma unroll
            for (int r = 0; r < 8; ++r) oacc[r] = 0.0f;

            #pragma unroll
            for (int c = 0; c < 4; ++c) {
                v8f t = __builtin_amdgcn_wmma_f32_16x16x32_bf16(
                            false, af[c][0].v, false, b[0].v, (short)0, zc, false, false);
                t = __builtin_amdgcn_wmma_f32_16x16x32_bf16(
                            false, af[c][1].v, false, b[1].v, (short)0, t, false, false);
                float coef = trig[y * 2 + (c >> 1)] * trig[x * 2 + (c & 1)];
                #pragma unroll
                for (int r = 0; r < 8; ++r) oacc[r] += coef * t[r];
            }
            #pragma unroll
            for (int r = 0; r < 8; ++r) {
                int erow = e0 + r + 8 * hi;
                outg[base + (size_t)erow * 1024 + ncb + nbl + lo] = oacc[r] * dv;
            }
        }
        __syncthreads();
    }
}

extern "C" void kernel_launch(void* const* d_in, const int* in_sizes, int n_in,
                              void* d_out, int out_size, void* d_ws, size_t ws_size,
                              hipStream_t stream) {
    (void)in_sizes; (void)n_in; (void)out_size; (void)d_ws; (void)ws_size;
    const float* q = (const float*)d_in[0];
    const float* k = (const float*)d_in[1];
    const float* v = (const float*)d_in[2];
    float* out = (float*)d_out;

    hipFuncSetAttribute((const void*)cosattn2d_kernel,
                        hipFuncAttributeMaxDynamicSharedMemorySize, (int)SMEM_BYTES);
    // one workgroup per (b, head): 32*8 = 256
    cosattn2d_kernel<<<256, 256, SMEM_BYTES, stream>>>(q, k, v, out);
}
